// RCNNNet_77884936946325
// MI455X (gfx1250) — matrix operand order repaired and checked by reading, not compiled, not run on hardware
//
#include <hip/hip_runtime.h>

// ---------------------------------------------------------------------------
// RCNN head (PointRCNN-style) for gfx1250.  GEMM-shaped layers run on
// v_wmma_f32_16x16x32_f16 (f16 in, f32 accumulate).  Weights are pre-swizzled
// on-device into WMMA B-fragment order so each wave loads its B operand with
// two global_load_b128.  A operands are staged in LDS as f16 row-major tiles.
// First SA layers use a K-row permutation ([feat | dxyz | pad]) so grouped
// feature rows can be gathered with aligned b128 copies.
// ---------------------------------------------------------------------------

typedef __attribute__((ext_vector_type(16))) _Float16 v16h;
typedef __attribute__((ext_vector_type(8)))  float    v8f;
typedef __attribute__((ext_vector_type(4)))  int      v4i;

#define BATCH 512
#define NPTS  512
#define NPT1  128
#define NPT2  32
#define NSAMP 64

// ---------------------------------------------------------------------------
// WMMA helpers
// ---------------------------------------------------------------------------
__device__ __forceinline__ v8f wmma32(v16h a, v16h b, v8f c) {
  return __builtin_amdgcn_wmma_f32_16x16x32_f16(
      /*neg_a=*/false, a, /*neg_b=*/false, b,
      /*c_mod=*/(short)0, c, /*reuse_a=*/false, /*reuse_b=*/false);
}

// A-matrix 16x32 f16 fragment from row-major LDS tile (stride ld halves).
// ISA layout (CDNA5 ISA): lanes 0-15 hold rows M=0..15 with K={0..7,16..23},
// lanes 16-31 hold K={8..15,24..31}.  Each half is 8 consecutive halves ->
// two ds_load_b128 per fragment.
__device__ __forceinline__ v16h lds_load_a(const _Float16* sm, int ld,
                                           int row0, int k0) {
  int lane = threadIdx.x & 31;
  int m  = row0 + (lane & 15);
  int kb = (lane >> 4) * 8;
  v16h a;
#pragma unroll
  for (int i = 0; i < 8; ++i) {
    int kl = (i < 4) ? (kb + 2 * i) : (16 + kb + 2 * (i - 4));
    a[2 * i]     = sm[(size_t)m * ld + k0 + kl];
    a[2 * i + 1] = sm[(size_t)m * ld + k0 + kl + 1];
  }
  return a;
}

// B fragments are pre-swizzled in global memory: tile (nt,kt) is 32 lanes x
// 16 halves contiguous.  One 32-byte vector load per fragment.
__device__ __forceinline__ v16h load_bfrag(const _Float16* wp, int nkt,
                                           int nt, int kt) {
  int lane = threadIdx.x & 31;
  return *(const v16h*)(wp + (((size_t)nt * nkt + kt) * 32 + lane) * 16);
}

// C/D 16x16 f32 layout: element r -> row = row0 + r + 8*(lane>=16),
// col = n0 + (lane&15).  Bias + ReLU epilogue into f16 LDS.
__device__ __forceinline__ void store_d_relu(_Float16* sm, int ld, int row0,
                                             int n0, v8f acc,
                                             const float* bias) {
  int lane = threadIdx.x & 31;
  int n = n0 + (lane & 15);
  float bv = bias[n];
  int mbase = row0 + ((lane >> 4) << 3);
#pragma unroll
  for (int r = 0; r < 8; ++r) {
    float v = acc[r] + bv;
    sm[(size_t)(mbase + r) * ld + n] = (_Float16)(v > 0.f ? v : 0.f);
  }
}

// ---------------------------------------------------------------------------
// Weight pre-swizzle: W (K x N fp32, row-major) -> f16 B-fragments.
// B-matrix 32x16 f16 striping: lane<16 holds K=0..7,16..23 (2/VGPR), lane>=16 holds
// K=8..15,24..31; col N = lane&15.  Rows k>=K are zero-padded.
// kmode=1 permutes K rows: logical k<kfeat reads source row k+(K-kfeat)
// (feature rows), logical k>=kfeat reads source row k-kfeat (xyz rows) --
// i.e. input layout becomes [feat | xyz | pad].
// ---------------------------------------------------------------------------
__global__ __launch_bounds__(512) void pack_weight_kernel(
    const float* __restrict__ W, _Float16* __restrict__ dst, int K, int Nn,
    int NKT, int kmode, int kfeat) {
  int tile = blockIdx.x;            // tile = nt*NKT + kt
  int nt = tile / NKT, kt = tile - nt * NKT;
  int tid = threadIdx.x;            // 512 = 32 lanes x 16 elements
  int lane = tid >> 4, j = tid & 15;
  int n = nt * 16 + (lane & 15);
  int k = kt * 32 + ((lane >> 4) << 3) + (j & 7) + ((j >> 3) << 4);
  float v = 0.f;
  if (k < K) {
    int src = k;
    if (kmode == 1) src = (k < kfeat) ? (k + (K - kfeat)) : (k - kfeat);
    v = W[(size_t)src * Nn + n];
  }
  dst[((size_t)tile * 32 + lane) * 16 + j] = (_Float16)v;
}

// ---------------------------------------------------------------------------
// Stage 1: fused per-point MLPs  (xyz_up 5->128->128, merge 256->128)
// 16 points per block, 4 waves.  Also exports xyz (fp32) and feat0 (f16).
// ---------------------------------------------------------------------------
__global__ __launch_bounds__(128) void pointwise_kernel(
    const float* __restrict__ pts,
    const _Float16* wpx1, const float* bx1,
    const _Float16* wpx2, const float* bx2,
    const _Float16* wpm,  const float* bm,
    float* __restrict__ xyz, _Float16* __restrict__ feat0) {
  int pt0 = blockIdx.x * 16;
  __shared__ __align__(16) _Float16 in5[16 * 32];
  __shared__ __align__(16) _Float16 rpn[16 * 128];
  __shared__ __align__(16) _Float16 x1 [16 * 128];
  __shared__ __align__(16) _Float16 x2 [16 * 128];
  int tid = threadIdx.x;
  int wu = __builtin_amdgcn_readfirstlane(tid) >> 5;  // uniform wave id

  for (int idx = tid; idx < 16 * 32; idx += 128) in5[idx] = (_Float16)0.f;
  __syncthreads();
  for (int idx = tid; idx < 16 * 133; idx += 128) {
    int p = idx / 133, ch = idx - p * 133;
    float v = pts[((size_t)pt0 + p) * 133 + ch];
    if (ch < 3) xyz[((size_t)pt0 + p) * 3 + ch] = v;
    if (ch < 5) in5[p * 32 + ch] = (_Float16)v;
    else        rpn[p * 128 + (ch - 5)] = (_Float16)v;
  }
  __syncthreads();

  // L1: K=32 (5 real + zero pad), N=128
  for (int nt = wu; nt < 8; nt += 4) {
    v8f acc = {};
    acc = wmma32(lds_load_a(in5, 32, 0, 0), load_bfrag(wpx1, 1, nt, 0), acc);
    store_d_relu(x1, 128, 0, nt * 16, acc, bx1);
  }
  __syncthreads();
  // L2: K=128
  for (int nt = wu; nt < 8; nt += 4) {
    v8f acc = {};
#pragma unroll
    for (int kt = 0; kt < 4; ++kt)
      acc = wmma32(lds_load_a(x1, 128, 0, kt * 32),
                   load_bfrag(wpx2, 4, nt, kt), acc);
    store_d_relu(x2, 128, 0, nt * 16, acc, bx2);
  }
  __syncthreads();
  // merge: K=256 = [x2 | rpn]
  for (int nt = wu; nt < 8; nt += 4) {
    v8f acc = {};
#pragma unroll
    for (int kt = 0; kt < 4; ++kt)
      acc = wmma32(lds_load_a(x2, 128, 0, kt * 32),
                   load_bfrag(wpm, 8, nt, kt), acc);
#pragma unroll
    for (int kt = 0; kt < 4; ++kt)
      acc = wmma32(lds_load_a(rpn, 128, 0, kt * 32),
                   load_bfrag(wpm, 8, nt, kt + 4), acc);
    int lane = tid & 31;
    int ncol = nt * 16 + (lane & 15);
    float bv = bm[ncol];
    int mbase = (lane >> 4) << 3;
#pragma unroll
    for (int r = 0; r < 8; ++r) {
      float v = acc[r] + bv;
      feat0[((size_t)pt0 + mbase + r) * 128 + ncol] =
          (_Float16)(v > 0.f ? v : 0.f);
    }
  }
}

// ---------------------------------------------------------------------------
// Farthest point sampling: 1 block per batch, iterative argmax (first-index
// tie-break, matching jnp.argmax).  Writes only new_xyz.
// ---------------------------------------------------------------------------
__global__ __launch_bounds__(256) void fps_kernel(const float* __restrict__ xyz,
                                                  int n, int np,
                                                  float* __restrict__ nxyz) {
  int b = blockIdx.x;
  const float* px = xyz + (size_t)b * n * 3;
  __shared__ float dists[512];
  __shared__ float rv[256];
  __shared__ int   ri[256];
  __shared__ int   s_last;
  int tid = threadIdx.x;
  for (int p = tid; p < n; p += 256) dists[p] = 1e10f;
  if (tid == 0) s_last = 0;
  if (tid < 3) nxyz[(size_t)(b * np) * 3 + tid] = px[tid];
  __syncthreads();
  for (int it = 1; it < np; ++it) {
    int last = s_last;
    float lx = px[last * 3], ly = px[last * 3 + 1], lz = px[last * 3 + 2];
    float bv = -1.f; int bi = 0;
    for (int p = tid; p < n; p += 256) {
      float dx = px[p * 3] - lx, dy = px[p * 3 + 1] - ly, dz = px[p * 3 + 2] - lz;
      float dd = fminf(dists[p], dx * dx + dy * dy + dz * dz);
      dists[p] = dd;
      if (dd > bv) { bv = dd; bi = p; }  // strict > keeps first index
    }
    rv[tid] = bv; ri[tid] = bi;
    __syncthreads();
    for (int s = 128; s > 0; s >>= 1) {
      if (tid < s) {
        float ov = rv[tid + s]; int oi = ri[tid + s];
        if (ov > rv[tid] || (ov == rv[tid] && oi < ri[tid])) {
          rv[tid] = ov; ri[tid] = oi;
        }
      }
      __syncthreads();
    }
    if (tid == 0) s_last = ri[0];
    __syncthreads();
    if (tid < 3) nxyz[((size_t)(b * np) + it) * 3 + tid] = px[s_last * 3 + tid];
    __syncthreads();
  }
}

// ---------------------------------------------------------------------------
// Ball query: one wave per center; take the FIRST nsample in-radius indices
// in ascending order (== sort(where(d2<=r2, idx, n))[:, :64]); pad with the
// first neighbor.
// ---------------------------------------------------------------------------
__global__ __launch_bounds__(32) void ballquery_kernel(
    const float* __restrict__ xyz, const float* __restrict__ nxyz, int n,
    int np, float r2, int* __restrict__ nbr) {
  int b = blockIdx.x / np, c = blockIdx.x - b * np;
  int lane = threadIdx.x;
  const float* px = xyz + (size_t)b * n * 3;
  const float* pc = nxyz + ((size_t)b * np + c) * 3;
  float cx = pc[0], cy = pc[1], cz = pc[2];
  int* out = nbr + ((size_t)b * np + c) * NSAMP;
  int count = 0, firstIdx = -1;
  for (int base = 0; base < n && count < NSAMP; base += 32) {
    int p = base + lane;
    bool ok = false;
    if (p < n) {
      float dx = px[p * 3] - cx, dy = px[p * 3 + 1] - cy, dz = px[p * 3 + 2] - cz;
      ok = (dx * dx + dy * dy + dz * dz) <= r2;
    }
    unsigned mm = (unsigned)__ballot(ok);
    if (firstIdx < 0 && mm != 0u) firstIdx = base + __builtin_ctz(mm);
    if (ok) {
      int pos = count + __popc(mm & ((1u << lane) - 1u));
      if (pos < NSAMP) out[pos] = p;
    }
    count += __popc(mm);
  }
  if (count > NSAMP) count = NSAMP;
  if (firstIdx < 0) firstIdx = n - 1;  // degenerate path of the reference
  for (int pos = count + lane; pos < NSAMP; pos += 32) out[pos] = firstIdx;
}

// ---------------------------------------------------------------------------
// Grouped 3-layer MLP + max-pool over 64 samples.  One center per block,
// 4 waves, each wave owns one 16-row M-tile of samples.
// Input layout per sample row (160 halves): [feat(128) | dxyz(3) | pad(29)]
// (layer-1 weights were K-permuted to match), so feature rows are gathered
// with aligned 16-byte copies.
// Layers: [160 -> 128] -> [128 -> 128] -> [128 -> N3].
// ---------------------------------------------------------------------------
__global__ __launch_bounds__(128) void sa_group_mlp_kernel(
    const float* __restrict__ xyz, const float* __restrict__ nxyz,
    const _Float16* __restrict__ feat, const int* __restrict__ nbr, int n,
    int np, const _Float16* wp1, const float* b1, const _Float16* wp2,
    const float* b2, const _Float16* wp3, const float* b3, int N3,
    _Float16* __restrict__ fout) {
  int b = blockIdx.x / np, c = blockIdx.x - b * np;
  __shared__ __align__(16) _Float16 g[64 * 160];
  __shared__ __align__(16) _Float16 h[64 * 256];
  __shared__ int nbs[64];
  int tid = threadIdx.x;
  int wu = __builtin_amdgcn_readfirstlane(tid) >> 5;  // uniform wave id
  if (tid < 64) nbs[tid] = nbr[((size_t)b * np + c) * NSAMP + tid];
  const float* pc = nxyz + ((size_t)b * np + c) * 3;
  float cx = pc[0], cy = pc[1], cz = pc[2];
  __syncthreads();
  // feature rows: 64 samples x 16 chunks of 16B (128 halves)
  for (int v = tid; v < 64 * 16; v += 128) {
    int s = v >> 4, ck = v & 15;
    int j = nbs[s];
    *(v4i*)(g + (size_t)s * 160 + ck * 8) =
        *(const v4i*)(feat + ((size_t)b * n + j) * 128 + ck * 8);
  }
  // dxyz + zero pad: halves 128..159 of each row
  for (int idx = tid; idx < 64 * 32; idx += 128) {
    int s = idx >> 5, ch = idx & 31;
    _Float16 val = (_Float16)0.f;
    if (ch < 3) {
      int j = nbs[s];
      float cv = (ch == 0) ? cx : (ch == 1) ? cy : cz;
      val = (_Float16)(xyz[((size_t)b * n + j) * 3 + ch] - cv);
    }
    g[(size_t)s * 160 + 128 + ch] = val;
  }
  __syncthreads();
  int row0 = wu * 16;
  // L1: K=160, N=128
  for (int nt = 0; nt < 8; ++nt) {
    v8f acc = {};
#pragma unroll
    for (int kt = 0; kt < 5; ++kt)
      acc = wmma32(lds_load_a(g, 160, row0, kt * 32),
                   load_bfrag(wp1, 5, nt, kt), acc);
    store_d_relu(h, 128, row0, nt * 16, acc, b1);
  }
  __syncthreads();
  // L2: K=128 (h -> g reused with ld=128)
  for (int nt = 0; nt < 8; ++nt) {
    v8f acc = {};
#pragma unroll
    for (int kt = 0; kt < 4; ++kt)
      acc = wmma32(lds_load_a(h, 128, row0, kt * 32),
                   load_bfrag(wp2, 4, nt, kt), acc);
    store_d_relu(g, 128, row0, nt * 16, acc, b2);
  }
  __syncthreads();
  // L3: K=128, N=N3 (g -> h with ld=N3)
  int nt3 = N3 >> 4;
  for (int nt = 0; nt < nt3; ++nt) {
    v8f acc = {};
#pragma unroll
    for (int kt = 0; kt < 4; ++kt)
      acc = wmma32(lds_load_a(g, 128, row0, kt * 32),
                   load_bfrag(wp3, 4, nt, kt), acc);
    store_d_relu(h, N3, row0, nt * 16, acc, b3);
  }
  __syncthreads();
  // max over 64 samples
  for (int col = tid; col < N3; col += 128) {
    float m = (float)h[col];
    for (int s = 1; s < 64; ++s) m = fmaxf(m, (float)h[(size_t)s * N3 + col]);
    fout[((size_t)b * np + c) * N3 + col] = (_Float16)m;
  }
}

// ---------------------------------------------------------------------------
// SA3: one batch per block.  Input row layout (288 halves):
// [f2(256) | xyz2(3) | pad(29)] (layer-1 weights K-permuted to match).
// Layers 288->256->256->512, max over 32 points -> f3 (fp32).
// ---------------------------------------------------------------------------
__global__ __launch_bounds__(128) void sa3_kernel(
    const float* __restrict__ xyz2, const _Float16* __restrict__ f2,
    const _Float16* wp1, const float* b1, const _Float16* wp2, const float* b2,
    const _Float16* wp3, const float* b3, float* __restrict__ f3) {
  int b = blockIdx.x;
  __shared__ __align__(16) _Float16 g[32 * 288];
  __shared__ __align__(16) _Float16 h[32 * 512];
  int tid = threadIdx.x;
  int wu = __builtin_amdgcn_readfirstlane(tid) >> 5;  // uniform wave id
  // f2 rows: 32 samples x 32 chunks of 16B (256 halves)
  for (int v = tid; v < 32 * 32; v += 128) {
    int s = v >> 5, ck = v & 31;
    *(v4i*)(g + (size_t)s * 288 + ck * 8) =
        *(const v4i*)(f2 + ((size_t)b * 32 + s) * 256 + ck * 8);
  }
  // xyz + zero pad: halves 256..287
  for (int idx = tid; idx < 32 * 32; idx += 128) {
    int s = idx >> 5, ch = idx & 31;
    _Float16 val = (_Float16)0.f;
    if (ch < 3) val = (_Float16)xyz2[((size_t)b * 32 + s) * 3 + ch];
    g[(size_t)s * 288 + 256 + ch] = val;
  }
  __syncthreads();
  // L1: M=32 (2 mtiles) x N=256 (16 ntiles), K=288
  for (int t = wu; t < 32; t += 4) {
    int mt = t & 1, nt = t >> 1;
    v8f acc = {};
#pragma unroll
    for (int kt = 0; kt < 9; ++kt)
      acc = wmma32(lds_load_a(g, 288, mt * 16, kt * 32),
                   load_bfrag(wp1, 9, nt, kt), acc);
    store_d_relu(h, 256, mt * 16, nt * 16, acc, b1);
  }
  __syncthreads();
  // L2: K=256 (h -> g with ld=256; fits in 32*288)
  for (int t = wu; t < 32; t += 4) {
    int mt = t & 1, nt = t >> 1;
    v8f acc = {};
#pragma unroll
    for (int kt = 0; kt < 8; ++kt)
      acc = wmma32(lds_load_a(h, 256, mt * 16, kt * 32),
                   load_bfrag(wp2, 8, nt, kt), acc);
    store_d_relu(g, 256, mt * 16, nt * 16, acc, b2);
  }
  __syncthreads();
  // L3: N=512 (32 ntiles), K=256 (g -> h with ld=512)
  for (int t = wu; t < 64; t += 4) {
    int mt = t & 1, nt = t >> 1;
    v8f acc = {};
#pragma unroll
    for (int kt = 0; kt < 8; ++kt)
      acc = wmma32(lds_load_a(g, 256, mt * 16, kt * 32),
                   load_bfrag(wp3, 8, nt, kt), acc);
    store_d_relu(h, 512, mt * 16, nt * 16, acc, b3);
  }
  __syncthreads();
  for (int col = tid; col < 512; col += 128) {
    float m = (float)h[col];
    for (int s = 1; s < 32; ++s) m = fmaxf(m, (float)h[(size_t)s * 512 + col]);
    f3[(size_t)b * 512 + col] = m;
  }
}

// ---------------------------------------------------------------------------
// Heads: tiny fp32 MLPs, one batch per block of 256 threads.
// out = [cls(512x1) | reg(512x46)]
// ---------------------------------------------------------------------------
__global__ __launch_bounds__(256) void heads_kernel(
    const float* __restrict__ f3, const float* cw1, const float* cb1,
    const float* cw2, const float* cb2, const float* cw3, const float* cb3,
    const float* rw1, const float* rb1, const float* rw2, const float* rb2,
    const float* rw3, const float* rb3, float* __restrict__ out) {
  int b = blockIdx.x, tid = threadIdx.x;
  __shared__ float x[512], h1[256], h2[256];
  x[tid]       = f3[(size_t)b * 512 + tid];
  x[tid + 256] = f3[(size_t)b * 512 + tid + 256];
  __syncthreads();
  // cls head
  {
    float s = cb1[tid];
    for (int k = 0; k < 512; ++k) s += x[k] * cw1[(size_t)k * 256 + tid];
    h1[tid] = s > 0.f ? s : 0.f;
    __syncthreads();
    float s2 = cb2[tid];
    for (int k = 0; k < 256; ++k) s2 += h1[k] * cw2[(size_t)k * 256 + tid];
    h2[tid] = s2 > 0.f ? s2 : 0.f;
    __syncthreads();
    if (tid == 0) {
      float s3 = cb3[0];
      for (int k = 0; k < 256; ++k) s3 += h2[k] * cw3[k];
      out[b] = s3;
    }
    __syncthreads();
  }
  // reg head
  {
    float s = rb1[tid];
    for (int k = 0; k < 512; ++k) s += x[k] * rw1[(size_t)k * 256 + tid];
    h1[tid] = s > 0.f ? s : 0.f;
    __syncthreads();
    float s2 = rb2[tid];
    for (int k = 0; k < 256; ++k) s2 += h1[k] * rw2[(size_t)k * 256 + tid];
    h2[tid] = s2 > 0.f ? s2 : 0.f;
    __syncthreads();
    if (tid < 46) {
      float s3 = rb3[tid];
      for (int k = 0; k < 256; ++k) s3 += h2[k] * rw3[(size_t)k * 46 + tid];
      out[512 + (size_t)b * 46 + tid] = s3;
    }
  }
}

// ---------------------------------------------------------------------------
// Host-side orchestration
// ---------------------------------------------------------------------------
extern "C" void kernel_launch(void* const* d_in, const int* in_sizes, int n_in,
                              void* d_out, int out_size, void* d_ws,
                              size_t ws_size, hipStream_t stream) {
  // d_in layout (setup_inputs() insertion order, params leaves flattened as
  // (W,b) pairs): 0 pts; 1-4 xyz_up; 5-6 merge; 7-12 sa1; 13-18 sa2;
  // 19-24 sa3; 25-30 cls; 31-36 reg.
  const float* pts = (const float*)d_in[0];

  // ---- workspace carve-up -------------------------------------------------
  char* base = (char*)d_ws;
  size_t off = 0;
  auto alloc = [&](size_t bytes) {
    off = (off + 255) & ~(size_t)255;
    void* p = base + off;
    off += bytes;
    return p;
  };
  _Float16* wbase = (_Float16*)alloc(446464 * sizeof(_Float16));  // packed W
  float*    xyz   = (float*)alloc((size_t)BATCH * NPTS * 3 * 4);
  _Float16* feat0 = (_Float16*)alloc((size_t)BATCH * NPTS * 128 * 2);
  float*    nxyz1 = (float*)alloc((size_t)BATCH * NPT1 * 3 * 4);
  int*      nbr1  = (int*)alloc((size_t)BATCH * NPT1 * NSAMP * 4);
  _Float16* f1    = (_Float16*)alloc((size_t)BATCH * NPT1 * 128 * 2);
  float*    nxyz2 = (float*)alloc((size_t)BATCH * NPT2 * 3 * 4);
  int*      nbr2  = (int*)alloc((size_t)BATCH * NPT2 * NSAMP * 4);
  _Float16* f2    = (_Float16*)alloc((size_t)BATCH * NPT2 * 256 * 2);
  float*    f3    = (float*)alloc((size_t)BATCH * 512 * 4);
  (void)ws_size; (void)n_in; (void)in_sizes; (void)out_size;

  // ---- pack the 12 WMMA weight matrices into fragment order ---------------
  // kmode=1 for the first layer of each SA stage: reorder K rows to
  // [feat | xyz | pad] so the gather can use aligned b128 copies.
  struct PackDesc { int widx, K, Kp, Nn, kmode, kfeat; };
  static const PackDesc pk[12] = {
      {1, 5, 32, 128, 0, 0},     {3, 128, 128, 128, 0, 0},
      {5, 256, 256, 128, 0, 0},
      {7, 131, 160, 128, 1, 128},{9, 128, 128, 128, 0, 0},
      {11, 128, 128, 128, 0, 0},
      {13, 131, 160, 128, 1, 128},{15, 128, 128, 128, 0, 0},
      {17, 128, 128, 256, 0, 0},
      {19, 259, 288, 256, 1, 256},{21, 256, 256, 256, 0, 0},
      {23, 256, 256, 512, 0, 0}};
  _Float16* wp[12];
  size_t hoff = 0;
  for (int i = 0; i < 12; ++i) {
    wp[i] = wbase + hoff;
    int NKT = pk[i].Kp / 32, NNT = pk[i].Nn / 16;
    pack_weight_kernel<<<NKT * NNT, 512, 0, stream>>>(
        (const float*)d_in[pk[i].widx], wp[i], pk[i].K, pk[i].Nn, NKT,
        pk[i].kmode, pk[i].kfeat);
    hoff += (size_t)pk[i].Kp * pk[i].Nn;
  }

  // ---- stage 1: per-point MLPs -------------------------------------------
  pointwise_kernel<<<(BATCH * NPTS) / 16, 128, 0, stream>>>(
      pts, wp[0], (const float*)d_in[2], wp[1], (const float*)d_in[4], wp[2],
      (const float*)d_in[6], xyz, feat0);

  // ---- SA1 ---------------------------------------------------------------
  fps_kernel<<<BATCH, 256, 0, stream>>>(xyz, NPTS, NPT1, nxyz1);
  ballquery_kernel<<<BATCH * NPT1, 32, 0, stream>>>(xyz, nxyz1, NPTS, NPT1,
                                                    0.04f, nbr1);
  sa_group_mlp_kernel<<<BATCH * NPT1, 128, 0, stream>>>(
      xyz, nxyz1, feat0, nbr1, NPTS, NPT1, wp[3], (const float*)d_in[8], wp[4],
      (const float*)d_in[10], wp[5], (const float*)d_in[12], 128, f1);

  // ---- SA2 ---------------------------------------------------------------
  fps_kernel<<<BATCH, 256, 0, stream>>>(nxyz1, NPT1, NPT2, nxyz2);
  ballquery_kernel<<<BATCH * NPT2, 32, 0, stream>>>(nxyz1, nxyz2, NPT1, NPT2,
                                                    0.16f, nbr2);
  sa_group_mlp_kernel<<<BATCH * NPT2, 128, 0, stream>>>(
      nxyz1, nxyz2, f1, nbr2, NPT1, NPT2, wp[6], (const float*)d_in[14], wp[7],
      (const float*)d_in[16], wp[8], (const float*)d_in[18], 256, f2);

  // ---- SA3 (global) ------------------------------------------------------
  sa3_kernel<<<BATCH, 128, 0, stream>>>(
      nxyz2, f2, wp[9], (const float*)d_in[20], wp[10], (const float*)d_in[22],
      wp[11], (const float*)d_in[24], f3);

  // ---- heads -------------------------------------------------------------
  heads_kernel<<<BATCH, 256, 0, stream>>>(
      f3, (const float*)d_in[25], (const float*)d_in[26],
      (const float*)d_in[27], (const float*)d_in[28], (const float*)d_in[29],
      (const float*)d_in[30], (const float*)d_in[31], (const float*)d_in[32],
      (const float*)d_in[33], (const float*)d_in[34], (const float*)d_in[35],
      (const float*)d_in[36], (float*)d_out);
}
